// Decoder6_73967926771747
// MI455X (gfx1250) — matrix operand order repaired and checked by reading, not compiled, not run on hardware
//
#include <hip/hip_runtime.h>
#include <hip/hip_bf16.h>
#include <stdint.h>

#define NTOK  3136
#define CDIM  128
#define BATCH 8
#define HEADS 2
#define DHEAD 64

typedef __attribute__((ext_vector_type(16))) _Float16 v16h;
typedef __attribute__((ext_vector_type(8)))  _Float16 v8h;
typedef __attribute__((ext_vector_type(8)))  float    v8f;

#define WMMA(a, b, c) \
  __builtin_amdgcn_wmma_f32_16x16x32_f16(false, (a), false, (b), (short)0, (c), false, false)

// ---- CDNA5 async global->LDS (ASYNCcnt-tracked, bypasses VGPRs) -----------
// Generic pointers to LDS carry the LDS byte offset in the low 32 bits
// (ISA 10.2 aperture); GV mode: vdst = LDS addr VGPR,
// vaddr = 64-bit global address pair.
__device__ __forceinline__ void async_load_lds_b128(void* lds_ptr, const void* gptr) {
  unsigned lds_off = (unsigned)(uintptr_t)lds_ptr;
  unsigned long long ga = (unsigned long long)(uintptr_t)gptr;
  asm volatile("global_load_async_to_lds_b128 %0, %1, off"
               :: "v"(lds_off), "v"(ga) : "memory");
}
__device__ __forceinline__ void wait_asynccnt0() {
  asm volatile("s_wait_asynccnt 0x0" ::: "memory");
}

// A-fragment (16x32 f16): lane holds row (lane&15); k = (i>=8?16:0) + (lane>=16?8:0) + (i&7).
__device__ __forceinline__ v16h frag_a_f16(const _Float16* p, int lane) {
  const int b0 = (lane & 16) ? 8 : 0;
  v8h lo = *(const v8h*)(p + b0);
  v8h hi = *(const v8h*)(p + b0 + 16);
  v16h r;
#pragma unroll
  for (int i = 0; i < 8; ++i) { r[i] = lo[i]; r[i + 8] = hi[i]; }
  return r;
}

// Same A-fragment but sourced from fp32 memory with on-the-fly convert.
__device__ __forceinline__ v16h frag_a_f32(const float* p, int lane) {
  const int b0 = (lane & 16) ? 8 : 0;
  v16h r;
#pragma unroll
  for (int i = 0; i < 8; ++i) {
    r[i]     = (_Float16)p[b0 + i];
    r[i + 8] = (_Float16)p[b0 + 16 + i];
  }
  return r;
}

// B-fragment (32x16 f16): lane holds column (lane&15); k = (i>=8?16:0) + (lane>=16?8:0) + (i&7).
__device__ __forceinline__ v16h frag_b_f16(const _Float16* p, int lane) {
  const int b0 = (lane & 16) ? 8 : 0;
  v8h lo = *(const v8h*)(p + b0);
  v8h hi = *(const v8h*)(p + b0 + 16);
  v16h r;
#pragma unroll
  for (int i = 0; i < 8; ++i) { r[i] = lo[i]; r[i + 8] = hi[i]; }
  return r;
}

// ---------------------------------------------------------------------------
// Kernel 0: convert the four 128x128 fp32 weight matrices to f16 (L2-resident).
// ---------------------------------------------------------------------------
__global__ void cvt_weights(const float* __restrict__ wq, const float* __restrict__ wk,
                            const float* __restrict__ wv, const float* __restrict__ wp,
                            _Float16* __restrict__ o) {
  int i = blockIdx.x * blockDim.x + threadIdx.x;
  if (i < CDIM * CDIM) {
    o[i]                   = (_Float16)wq[i];
    o[CDIM * CDIM + i]     = (_Float16)wk[i];
    o[2 * CDIM * CDIM + i] = (_Float16)wv[i];
    o[3 * CDIM * CDIM + i] = (_Float16)wp[i];
  }
}

// ---------------------------------------------------------------------------
// Kernel 1: fused QKV projection. 4 waves/block, each wave: 16-row strip.
// out = x @ W.T + b, Q additionally scaled by 1/sqrt(dh). Stored [b][h][n][64] f16.
// ---------------------------------------------------------------------------
__global__ __launch_bounds__(128) void qkv_proj(
    const float* __restrict__ x, const _Float16* __restrict__ Wh,
    const float* __restrict__ bq, const float* __restrict__ bk, const float* __restrict__ bv,
    _Float16* __restrict__ Qo, _Float16* __restrict__ Ko, _Float16* __restrict__ Vo) {
  const int lane = threadIdx.x & 31, wave = threadIdx.x >> 5;
  const int hi8 = (lane & 16) ? 8 : 0;
  const long row0 = (long)blockIdx.x * 64 + wave * 16;   // strip start in [B*N]
  const int b  = (int)(row0 / NTOK);                     // strips never straddle b
  const int n0 = (int)(row0 % NTOK);

  const float* xr = x + (row0 + (lane & 15)) * CDIM;
  v16h a0 = frag_a_f32(xr, lane);
  v16h a1 = frag_a_f32(xr + 32, lane);
  v16h a2 = frag_a_f32(xr + 64, lane);
  v16h a3 = frag_a_f32(xr + 96, lane);

  const float* biases[3] = { bq, bk, bv };
  _Float16* outs[3]      = { Qo, Ko, Vo };

#pragma unroll 1
  for (int wsel = 0; wsel < 3; ++wsel) {
    const _Float16* Wm = Wh + wsel * CDIM * CDIM;  // row-major [n_out][k]
    const float post = (wsel == 0) ? 0.125f : 1.0f;
#pragma unroll 1
    for (int nt = 0; nt < 8; ++nt) {
      const int ncol = nt * 16 + (lane & 15);
      const _Float16* wr = Wm + (long)ncol * CDIM;  // B[k][ncol] = W[ncol][k], contiguous k
      v8f c = {};
      c = WMMA(a0, frag_b_f16(wr, lane), c);
      c = WMMA(a1, frag_b_f16(wr + 32, lane), c);
      c = WMMA(a2, frag_b_f16(wr + 64, lane), c);
      c = WMMA(a3, frag_b_f16(wr + 96, lane), c);
      const float bval = biases[wsel][ncol];
      const int head = ncol >> 6, dh = ncol & 63;
      _Float16* base = outs[wsel] + (long)(b * HEADS + head) * NTOK * DHEAD;
#pragma unroll
      for (int r = 0; r < 8; ++r)
        base[(long)(n0 + hi8 + r) * DHEAD + dh] = (_Float16)((c[r] + bval) * post);
    }
  }
}

// ---------------------------------------------------------------------------
// Kernel 2: flash attention with async-to-LDS double buffering.
// Block = (64 queries, head, batch), 4 waves; wave w owns query rows [16w,16w+16).
// K tiles staged via GLOBAL_LOAD_ASYNC_TO_LDS_B128 (ASYNCcnt); V tiles via
// vector loads + transposed ds stores. Next tile issued before current compute.
// ---------------------------------------------------------------------------
__global__ __launch_bounds__(128) void flash_attn(
    const _Float16* __restrict__ Q, const _Float16* __restrict__ K,
    const _Float16* __restrict__ V, const float* __restrict__ rp,
    _Float16* __restrict__ AT) {
  __shared__ __align__(16) _Float16 sQ[64][72];      // [query][dh]
  __shared__ __align__(16) _Float16 sK[2][32][72];   // [buf][key][dh]
  __shared__ __align__(16) _Float16 sVT[2][64][40];  // [buf][dh][key] (transposed)
  __shared__ __align__(16) _Float16 sP[64][40];      // [query][key] probs
  __shared__ __align__(16) float    sS[64][36];      // [query][key] scores
  __shared__ float rowmax[64], rowsum[64], rowscale[64];

  const int tid = threadIdx.x;
  const int lane = tid & 31, wave = tid >> 5;
  const int hi8 = (lane & 16) ? 8 : 0;
  const int qt = blockIdx.x, h = blockIdx.y, b = blockIdx.z;
  const int q0 = qt * 64;
  const long bh = (long)(b * HEADS + h);
  const _Float16* Qg = Q + bh * NTOK * DHEAD;
  const _Float16* Kg = K + bh * NTOK * DHEAD;
  const _Float16* Vg = V + bh * NTOK * DHEAD;
  const float* rpb = rp + (long)h * NTOK * NTOK;

  // ---- prologue: async-stage Q tile + K tile 0; stage V tile 0 ------------
  for (int c = tid; c < 64 * 8; c += 128) {          // Q: 64x64 f16, 512 x 16B
    int r = c >> 3, cc = (c & 7) * 8;
    async_load_lds_b128(&sQ[r][cc], Qg + (long)(q0 + r) * DHEAD + cc);
  }
  for (int c = tid; c < 32 * 8; c += 128) {          // K tile 0: 256 x 16B
    int r = c >> 3, cc = (c & 7) * 8;
    async_load_lds_b128(&sK[0][r][cc], Kg + (long)r * DHEAD + cc);
  }
  for (int c = tid; c < 32 * 8; c += 128) {          // V tile 0: vector load + transpose
    int kr = c >> 3, d0 = (c & 7) * 8;
    v8h val = *(const v8h*)(Vg + (long)kr * DHEAD + d0);
#pragma unroll
    for (int j = 0; j < 8; ++j) sVT[0][d0 + j][kr] = val[j];
  }
  if (tid < 64) { rowmax[tid] = -1e30f; rowsum[tid] = 0.0f; }
  wait_asynccnt0();
  __syncthreads();

  const _Float16* qr = &sQ[wave * 16 + (lane & 15)][0];
  v16h aQ0 = frag_a_f16(qr, lane);
  v16h aQ1 = frag_a_f16(qr + 32, lane);

  v8f o0 = {}, o1 = {}, o2 = {}, o3 = {};

  const int KT = NTOK / 32;                          // 98 key tiles
  for (int kt = 0; kt < KT; ++kt) {
    const int k0 = kt * 32;
    const int cb = kt & 1, nb = cb ^ 1;

    // ---- prefetch next tile into the other buffer (overlaps compute) ------
    if (kt + 1 < KT) {
      const int kn = k0 + 32;
      for (int c = tid; c < 32 * 8; c += 128) {
        int r = c >> 3, cc = (c & 7) * 8;
        async_load_lds_b128(&sK[nb][r][cc], Kg + (long)(kn + r) * DHEAD + cc);
      }
      for (int c = tid; c < 32 * 8; c += 128) {
        int kr = c >> 3, d0 = (c & 7) * 8;
        v8h val = *(const v8h*)(Vg + (long)(kn + kr) * DHEAD + d0);
#pragma unroll
        for (int j = 0; j < 8; ++j) sVT[nb][d0 + j][kr] = val[j];
      }
    }

    // ---- S = Q * K^T (two 16x16 key sub-tiles x two dh chunks) ------------
    v8f s0 = {}, s1 = {};
    {
      const _Float16* kr0 = &sK[cb][(lane & 15)][0];
      const _Float16* kr1 = &sK[cb][16 + (lane & 15)][0];
      s0 = WMMA(aQ0, frag_b_f16(kr0, lane), s0);
      s0 = WMMA(aQ1, frag_b_f16(kr0 + 32, lane), s0);
      s1 = WMMA(aQ0, frag_b_f16(kr1, lane), s1);
      s1 = WMMA(aQ1, frag_b_f16(kr1 + 32, lane), s1);
    }
    // ---- spill to LDS with streamed relative-position bias ----------------
#pragma unroll
    for (int r = 0; r < 8; ++r) {
      const int m = wave * 16 + hi8 + r;
      const long qrow = q0 + m;
      sS[m][lane & 15]        = s0[r] + rpb[qrow * NTOK + k0 + (lane & 15)];
      sS[m][16 + (lane & 15)] = s1[r] + rpb[qrow * NTOK + k0 + 16 + (lane & 15)];
      __builtin_prefetch(&rpb[qrow * NTOK + k0 + 32 + (lane & 15)], 0, 0);
    }
    __syncthreads();

    // ---- online softmax, one thread per query row -------------------------
    if (tid < 64) {
      float mo = rowmax[tid], tm = -1e30f;
#pragma unroll
      for (int j = 0; j < 32; ++j) tm = fmaxf(tm, sS[tid][j]);
      const float nm = fmaxf(mo, tm);
      const float sc = __expf(mo - nm);
      float sum = 0.0f;
#pragma unroll
      for (int j = 0; j < 32; ++j) {
        const float p = __expf(sS[tid][j] - nm);
        sP[tid][j] = (_Float16)p;
        sum += p;
      }
      rowmax[tid] = nm;
      rowsum[tid] = rowsum[tid] * sc + sum;
      rowscale[tid] = sc;
    }
    __syncthreads();

    // ---- rescale running O and accumulate P*V (one WMMA per d-tile) -------
#pragma unroll
    for (int r = 0; r < 8; ++r) {
      const float sc = rowscale[wave * 16 + hi8 + r];
      o0[r] *= sc; o1[r] *= sc; o2[r] *= sc; o3[r] *= sc;
    }
    v16h aP = frag_a_f16(&sP[wave * 16 + (lane & 15)][0], lane);
    o0 = WMMA(aP, frag_b_f16(&sVT[cb][0  + (lane & 15)][0], lane), o0);
    o1 = WMMA(aP, frag_b_f16(&sVT[cb][16 + (lane & 15)][0], lane), o1);
    o2 = WMMA(aP, frag_b_f16(&sVT[cb][32 + (lane & 15)][0], lane), o2);
    o3 = WMMA(aP, frag_b_f16(&sVT[cb][48 + (lane & 15)][0], lane), o3);

    wait_asynccnt0();          // next-tile async K writes complete
    __syncthreads();           // everyone done reading cb / writing nb
  }

  // ---- normalize and emit f16 attention output [b][n][h*64 + d] -----------
#pragma unroll
  for (int r = 0; r < 8; ++r) {
    const int m = wave * 16 + hi8 + r;
    const float inv = 1.0f / rowsum[m];
    _Float16* op = AT + ((long)b * NTOK + (q0 + m)) * CDIM + h * DHEAD + (lane & 15);
    op[0]  = (_Float16)(o0[r] * inv);
    op[16] = (_Float16)(o1[r] * inv);
    op[32] = (_Float16)(o2[r] * inv);
    op[48] = (_Float16)(o3[r] * inv);
  }
}

// ---------------------------------------------------------------------------
// Kernel 3: output projection, f16 WMMA -> fp32 d_out.
// ---------------------------------------------------------------------------
__global__ __launch_bounds__(128) void out_proj(
    const _Float16* __restrict__ AT, const _Float16* __restrict__ Wp,
    const float* __restrict__ bp, float* __restrict__ out) {
  const int lane = threadIdx.x & 31, wave = threadIdx.x >> 5;
  const int hi8 = (lane & 16) ? 8 : 0;
  const long row0 = (long)blockIdx.x * 64 + wave * 16;

  const _Float16* ar = AT + (row0 + (lane & 15)) * CDIM;
  v16h a0 = frag_a_f16(ar, lane);
  v16h a1 = frag_a_f16(ar + 32, lane);
  v16h a2 = frag_a_f16(ar + 64, lane);
  v16h a3 = frag_a_f16(ar + 96, lane);

#pragma unroll 1
  for (int nt = 0; nt < 8; ++nt) {
    const int ncol = nt * 16 + (lane & 15);
    const _Float16* wr = Wp + (long)ncol * CDIM;
    v8f c = {};
    c = WMMA(a0, frag_b_f16(wr, lane), c);
    c = WMMA(a1, frag_b_f16(wr + 32, lane), c);
    c = WMMA(a2, frag_b_f16(wr + 64, lane), c);
    c = WMMA(a3, frag_b_f16(wr + 96, lane), c);
    const float bval = bp[ncol];
#pragma unroll
    for (int r = 0; r < 8; ++r)
      out[(row0 + hi8 + r) * CDIM + ncol] = c[r] + bval;
  }
}

// ---------------------------------------------------------------------------
extern "C" void kernel_launch(void* const* d_in, const int* in_sizes, int n_in,
                              void* d_out, int out_size, void* d_ws, size_t ws_size,
                              hipStream_t stream) {
  const float* x  = (const float*)d_in[0];
  // d_in[1]=H, d_in[2]=W (unused on sr_ratio==1 path)
  const float* rp = (const float*)d_in[3];
  const float* Wq = (const float*)d_in[4];
  const float* bq = (const float*)d_in[5];
  const float* Wk = (const float*)d_in[6];
  const float* bk = (const float*)d_in[7];
  const float* Wv = (const float*)d_in[8];
  const float* bv = (const float*)d_in[9];
  const float* Wp = (const float*)d_in[10];
  const float* bp = (const float*)d_in[11];
  float* out = (float*)d_out;

  _Float16* wsh = (_Float16*)d_ws;
  _Float16* Wh = wsh;                                   // 4 x 128x128 f16 (q,k,v,p)
  const size_t QKVE = (size_t)BATCH * HEADS * NTOK * DHEAD;
  _Float16* Qb = wsh + 4 * CDIM * CDIM;
  _Float16* Kb = Qb + QKVE;
  _Float16* Vb = Kb + QKVE;
  _Float16* AT = Vb + QKVE;                             // [B][N][C] f16

  cvt_weights<<<(CDIM * CDIM + 255) / 256, 256, 0, stream>>>(Wq, Wk, Wv, Wp, Wh);

  const int rows = BATCH * NTOK;                        // 25088
  qkv_proj<<<rows / 64, 128, 0, stream>>>(x, Wh, bq, bk, bv, Qb, Kb, Vb);

  dim3 g2(NTOK / 64, HEADS, BATCH);                     // (49, 2, 8)
  flash_attn<<<g2, 128, 0, stream>>>(Qb, Kb, Vb, rp, AT);

  out_proj<<<rows / 64, 128, 0, stream>>>(AT, Wh + 3 * CDIM * CDIM, bp, out);
}
